// NonLocal_32925219291718
// MI455X (gfx1250) — matrix operand order repaired and checked by reading, not compiled, not run on hardware
//
#include <hip/hip_runtime.h>
#include <stdint.h>

// ---------------------------------------------------------------------------
// Non-local block, fused flash-attention implementation for gfx1250 (MI455X).
// All GEMMs run on v_wmma_f32_16x16x32_bf16 (wave32 WMMA, f32 accumulate).
// Problem: B=8, C=256, H=W=64 (N=4096), Ci=128.
// Workspace: Q | K | V | Y bf16 (8 MiB each) | bf16 weights (256 KiB).
// Q/K/Y tiles are staged by the Tensor Data Mover (tensor_load_to_lds,
// TENSORcnt).  B fragments are software double-buffered so global loads
// overlap the WMMA pipe instead of serializing on s_wait_loadcnt 0.
// ---------------------------------------------------------------------------

#define C_    256
#define CI    128
#define NPIX  4096
#define BATCH 8
#define BR    128
#define BC    32

typedef __attribute__((ext_vector_type(16))) __bf16 v16bf;
typedef __attribute__((ext_vector_type(8)))  float  v8f;

union Frag16 {
    v16bf v;
    uint4 q[2];
    unsigned short s[16];
};

#ifndef __has_builtin
#define __has_builtin(x) 0
#endif

#if __has_builtin(__builtin_amdgcn_cvt_pk_bf16_f32)
#define HAVE_PKBF 1
#else
#define HAVE_PKBF 0
#endif

__device__ __forceinline__ unsigned short f2bfu_sw(float f) {
    unsigned u = __float_as_uint(f);
    unsigned r = u + 0x7FFFu + ((u >> 16) & 1u);   // round-to-nearest-even
    return (unsigned short)(r >> 16);
}

// pack two f32 -> {bf16 lo, bf16 hi} in one u32
__device__ __forceinline__ unsigned pack2bf(float lo, float hi) {
#if HAVE_PKBF
    auto t = __builtin_amdgcn_cvt_pk_bf16_f32(lo, hi);
    unsigned u; __builtin_memcpy(&u, &t, 4); return u;
#else
    return (unsigned)f2bfu_sw(lo) | ((unsigned)f2bfu_sw(hi) << 16);
#endif
}

__device__ __forceinline__ unsigned short f2bfu(float f) {
#if HAVE_PKBF
    return (unsigned short)(pack2bf(f, 0.f) & 0xFFFFu);
#else
    return f2bfu_sw(f);
#endif
}

__device__ __forceinline__ v8f wmma_bf16(const Frag16& a, const Frag16& b, v8f c) {
    return __builtin_amdgcn_wmma_f32_16x16x32_bf16(false, a.v, false, b.v,
                                                   (short)0, c, false, false);
}

// A fragment (16x32 bf16, M x K): lane holds row M=lane&15.
__device__ __forceinline__ void ldA(Frag16& f, const unsigned short* row, int k0, int half) {
    const unsigned short* p = row + k0 + half * 16;
    f.q[0] = *(const uint4*)(p);
    f.q[1] = *(const uint4*)(p + 8);
}

// B fragment (32x16 bf16, K x N, sourced from an N x K row-major tile).
__device__ __forceinline__ void ldB(Frag16& f, const unsigned short* p) {
    f.q[0] = *(const uint4*)(p);
    f.q[1] = *(const uint4*)(p + 8);
}

// ------------------------- Tensor Data Mover support -----------------------
#if __has_builtin(__builtin_amdgcn_tensor_load_to_lds) && \
    __has_builtin(__builtin_amdgcn_s_wait_tensorcnt)
#define USE_TDM 1
#else
#define USE_TDM 0
#endif

#if USE_TDM
typedef unsigned int u32x4 __attribute__((ext_vector_type(4)));
typedef int          i32x8 __attribute__((ext_vector_type(8)));
typedef int          i32x4 __attribute__((ext_vector_type(4)));

// Load `nrows` rows of a row-major (NPIX x 128) bf16 tensor into LDS,
// padding 4 dwords (8 elems) after each 64-dword row so the destination
// matches the 136-element padded LDS tile stride.
__device__ __forceinline__ void tdm_load_rows(unsigned lds_byte_off,
                                              const unsigned short* gsrc,
                                              int nrows)
{
    unsigned long long ga = (unsigned long long)(size_t)gsrc;
    u32x4 g0;
    g0[0] = 1u;                                            // count=1 (valid user D#)
    g0[1] = lds_byte_off;                                  // lds_addr
    g0[2] = (unsigned)(ga & 0xFFFFFFFFu);                  // global_addr[31:0]
    g0[3] = (unsigned)((ga >> 32) & 0x01FFFFFFu)           // global_addr[56:32]
          | (2u << 30);                                    // type = 2 ("image")
    i32x8 g1;
    g1[0] = (int)((1u << 16)      // data_size = 1 -> 2 bytes
                | (1u << 20)      // pad_enable
                | (5u << 22)      // pad_interval: 64 dwords (one 256B row)
                | (3u << 25));    // pad_amount: 4 dwords (16B)
    g1[1] = (int)(128u << 16);    // tensor_dim0 = 128 (low 16 in [31:16])
    g1[2] = (int)(4096u << 16);   // dim0 hi = 0 | tensor_dim1 = 4096 (lo 16)
    g1[3] = (int)(128u << 16);    // dim1 hi = 0 | tile_dim0 = 128
    g1[4] = nrows & 0xFFFF;       // tile_dim1 = nrows, tile_dim2 = 0
    g1[5] = 128;                  // tensor_dim0_stride = 128 (low 32)
    g1[6] = 0;
    g1[7] = 0;
    i32x4 z4 = {0, 0, 0, 0};
#if __clang_major__ >= 23
    i32x8 z8 = {0, 0, 0, 0, 0, 0, 0, 0};
    __builtin_amdgcn_tensor_load_to_lds(g0, g1, z4, z4, z8, 0);
#else
    __builtin_amdgcn_tensor_load_to_lds(g0, g1, z4, z4, 0);
#endif
}
#endif // USE_TDM

// ---------------------------------------------------------------------------
// Kernel 0: one-time f32 -> bf16 conversion of all four weight matrices.
// Layout in wbuf: [w_theta | w_phi | w_g | w_out], 32768 elements each.
// ---------------------------------------------------------------------------
__global__ __launch_bounds__(256) void cvtw_kernel(
    const float* __restrict__ w0, const float* __restrict__ w1,
    const float* __restrict__ w2, const float* __restrict__ w3,
    unsigned int* __restrict__ out)
{
    int idx = blockIdx.x * 256 + threadIdx.x;      // 0 .. 65535 u32 pairs
    int seg = idx >> 14, loc = idx & 16383;
    const float* src = (seg == 0) ? w0 : (seg == 1) ? w1 : (seg == 2) ? w2 : w3;
    float2 f = *(const float2*)(src + 2 * loc);
    out[idx] = pack2bf(f.x, f.y);
}

// ---------------------------------------------------------------------------
// Kernel 1: theta/phi/g projections.  Q[b,n,ci] = sum_c x[b,c,n]*W[ci,c]+bias
// 64 pixel-rows per block, full Ci=128, K=C=256.  4 waves x 16 rows.
// B fragments double-buffered: loads for j+1 issue before the wait for j.
// ---------------------------------------------------------------------------
typedef unsigned short AsTile[C_ + 8];

__device__ __forceinline__ void proj_one(
    const AsTile* As, const unsigned short* __restrict__ w,
    const float* __restrict__ bias, unsigned short* __restrict__ out,
    int n0, int wave, int half, int l15)
{
    v8f acc[8];
    v8f zero8 = {0.f,0.f,0.f,0.f,0.f,0.f,0.f,0.f};
#pragma unroll
    for (int j = 0; j < 8; ++j) acc[j] = zero8;
    const unsigned short* wrow = w + (size_t)l15 * C_ + half * 16;
    for (int kc = 0; kc < C_; kc += 32) {
        Frag16 a; ldA(a, &As[16 * wave + l15][0], kc, half);
        Frag16 b0, b1;
        ldB(b0, wrow + kc);
#pragma unroll
        for (int j = 0; j < 8; ++j) {
            Frag16* cur = (j & 1) ? &b1 : &b0;
            Frag16* nxt = (j & 1) ? &b0 : &b1;
            if (j < 7) ldB(*nxt, wrow + (size_t)(16 * (j + 1)) * C_ + kc);
            acc[j] = wmma_bf16(a, *cur, acc[j]);
        }
    }
#pragma unroll
    for (int j = 0; j < 8; ++j) {
        int col = 16 * j + l15;
        float bv = bias[col];
#pragma unroll
        for (int v = 0; v < 8; ++v) {
            int row = n0 + 16 * wave + v + 8 * half;        // C/D layout row
            out[(size_t)row * CI + col] = f2bfu(acc[j][v] + bv);
        }
    }
}

__global__ __launch_bounds__(128) void proj_kernel(
    const float* __restrict__ x,
    const unsigned short* __restrict__ w_th, const float* __restrict__ b_th,
    const unsigned short* __restrict__ w_ph, const float* __restrict__ b_ph,
    const unsigned short* __restrict__ w_g,  const float* __restrict__ b_g,
    unsigned short* __restrict__ qbuf,
    unsigned short* __restrict__ kbuf,
    unsigned short* __restrict__ vbuf)
{
    __shared__ unsigned short As[64][C_ + 8];           // 64 x 256 bf16, stride 264

    const int b    = blockIdx.y;
    const int n0   = blockIdx.x * 64;
    const int tid  = threadIdx.x;
    const int wave = tid >> 5, lane = tid & 31;
    const int half = lane >> 4, l15 = lane & 15;
    const float* xb = x + (size_t)b * C_ * NPIX;

    // Stage As[n][c] = bf16(x[b, c, n0+n])  via float4 loads along n
    for (int i = 0; i < 32; ++i) {
        int e = i * 128 + tid;                  // 4096 float4 groups
        int qq = e & 15, c = e >> 4;
        const float4 xv = *(const float4*)(xb + (size_t)c * NPIX + n0 + 4 * qq);
        int n = 4 * qq;
        As[n    ][c] = f2bfu(xv.x);
        As[n + 1][c] = f2bfu(xv.y);
        As[n + 2][c] = f2bfu(xv.z);
        As[n + 3][c] = f2bfu(xv.w);
    }
    __syncthreads();

    const size_t bofs = (size_t)b * NPIX * CI;
    proj_one(As, w_th, b_th, qbuf + bofs, n0, wave, half, l15);
    proj_one(As, w_ph, b_ph, kbuf + bofs, n0, wave, half, l15);
    proj_one(As, w_g,  b_g,  vbuf + bofs, n0, wave, half, l15);
}

// ---------------------------------------------------------------------------
// Kernel 2: flash attention.  att = softmax(Q K^T) (no scale, per reference),
// Y = att V.  Br=128 rows/block (8 waves x 16 rows), Bc=32 keys/iteration.
// Total LDS: (128+32)*136 + 128*40 + 8*16*40 = 32000 elems = 64000 bytes.
// ---------------------------------------------------------------------------
#define QSTR (CI + 8)
#define VSTR (BC + 8)
#define QOFF 0
#define KOFF (BR * QSTR)
#define VOFF (KOFF + BC * QSTR)
#define POFF (VOFF + CI * VSTR)
#define LDSN (POFF + 8 * 16 * VSTR)

__global__ __launch_bounds__(256) void attn_kernel(
    const unsigned short* __restrict__ qbuf,
    const unsigned short* __restrict__ kbuf,
    const unsigned short* __restrict__ vbuf,
    unsigned short* __restrict__ ybuf)
{
    __shared__ unsigned short lds[LDSN];
    unsigned short* Qs = &lds[QOFF];   // [BR][QSTR]
    unsigned short* Ks = &lds[KOFF];   // [BC][QSTR]
    unsigned short* Vt = &lds[VOFF];   // [CI][VSTR]  (V transposed: ci x key)
    unsigned short* Ps = &lds[POFF];   // [8][16][VSTR] per-wave P buffer

    const int b    = blockIdx.y;
    const int q0   = blockIdx.x * BR;
    const int tid  = threadIdx.x;
    const int wave = tid >> 5, lane = tid & 31;
    const int half = lane >> 4, l15 = lane & 15;

    const unsigned short* qg = qbuf + (size_t)b * NPIX * CI;
    const unsigned short* kg = kbuf + (size_t)b * NPIX * CI;
    const unsigned int*   vg = (const unsigned int*)(vbuf + (size_t)b * NPIX * CI);

    // ---- stage Q tile (128 x 128 bf16) ----
#if USE_TDM
    if (wave == 0) {
        tdm_load_rows((unsigned)(QOFF * 2), qg + (size_t)q0 * CI, BR);
        __builtin_amdgcn_s_wait_tensorcnt(0);
    }
#else
    for (int i = 0; i < 32; ++i) {
        int e = i * 256 + tid; int pc = e & 63, r = e >> 6;
        *(unsigned int*)&Qs[r * QSTR + 2 * pc] =
            ((const unsigned int*)qg)[(size_t)(q0 + r) * 64 + pc];
    }
#endif

    v8f o[8];
    v8f zero8 = {0.f,0.f,0.f,0.f,0.f,0.f,0.f,0.f};
#pragma unroll
    for (int j = 0; j < 8; ++j) o[j] = zero8;
    float mrow[8], lrow[8];
#pragma unroll
    for (int v = 0; v < 8; ++v) { mrow[v] = -3.0e38f; lrow[v] = 0.f; }

    for (int m0 = 0; m0 < NPIX; m0 += BC) {
        __syncthreads();   // previous iteration's readers are done
#if USE_TDM
        if (wave == 0)
            tdm_load_rows((unsigned)(KOFF * 2), kg + (size_t)m0 * CI, BC);
#else
        for (int i = 0; i < 8; ++i) {
            int e = i * 256 + tid; int pc = e & 63, r = e >> 6;
            *(unsigned int*)&Ks[r * QSTR + 2 * pc] =
                ((const unsigned int*)kg)[(size_t)(m0 + r) * 64 + pc];
        }
#endif
        // Stage V transposed (ci x key): two b32 LDS stores per uint pair
        for (int i = 0; i < 4; ++i) {
            int e = i * 256 + tid; int pc = e & 63, rp = e >> 6;
            int r = 2 * rp;
            unsigned vw0 = vg[(size_t)(m0 + r    ) * 64 + pc];
            unsigned vw1 = vg[(size_t)(m0 + r + 1) * 64 + pc];
            unsigned lo = (vw0 & 0xFFFFu) | (vw1 << 16);          // ci, keys r..r+1
            unsigned hi = (vw0 >> 16) | (vw1 & 0xFFFF0000u);      // ci+1
            *(unsigned*)&Vt[(2 * pc    ) * VSTR + r] = lo;
            *(unsigned*)&Vt[(2 * pc + 1) * VSTR + r] = hi;
        }
#if USE_TDM
        if (wave == 0) __builtin_amdgcn_s_wait_tensorcnt(0);
#endif
        __syncthreads();

        // S = Q K^T : 16 x 32 per wave
        v8f s[2]; s[0] = zero8; s[1] = zero8;
        for (int kc = 0; kc < CI; kc += 32) {
            Frag16 a; ldA(a, &Qs[(16 * wave + l15) * QSTR], kc, half);
#pragma unroll
            for (int j = 0; j < 2; ++j) {
                Frag16 bk; ldB(bk, &Ks[(16 * j + l15) * QSTR + kc + half * 16]);
                s[j] = wmma_bf16(a, bk, s[j]);
            }
        }

        // Online softmax per row (a half's 16 lanes hold that row's columns)
#pragma unroll
        for (int v = 0; v < 8; ++v) {
            float mx = fmaxf(s[0][v], s[1][v]);
            for (int off = 1; off < 16; off <<= 1)
                mx = fmaxf(mx, __shfl_xor(mx, off, 32));
            float mn = fmaxf(mrow[v], mx);
            float sc = __expf(mrow[v] - mn);
            float p0 = __expf(s[0][v] - mn);
            float p1 = __expf(s[1][v] - mn);
            s[0][v] = p0; s[1][v] = p1;
            float rs = p0 + p1;
            for (int off = 1; off < 16; off <<= 1)
                rs += __shfl_xor(rs, off, 32);
            lrow[v] = lrow[v] * sc + rs;
            mrow[v] = mn;
#pragma unroll
            for (int j = 0; j < 8; ++j) o[j][v] *= sc;
        }

        // P: D-layout -> per-wave LDS -> A-layout (same-wave DS ops in order)
#pragma unroll
        for (int j = 0; j < 2; ++j)
#pragma unroll
            for (int v = 0; v < 8; ++v)
                Ps[(wave * 16 + v + 8 * half) * VSTR + 16 * j + l15] = f2bfu(s[j][v]);

        // O += P V  (V fragments double-buffered from LDS)
        Frag16 pa; ldA(pa, &Ps[(wave * 16 + l15) * VSTR], 0, half);
        {
            Frag16 v0, v1;
            ldB(v0, &Vt[(size_t)l15 * VSTR + half * 16]);
#pragma unroll
            for (int j = 0; j < 8; ++j) {
                Frag16* cur = (j & 1) ? &v1 : &v0;
                Frag16* nxt = (j & 1) ? &v0 : &v1;
                if (j < 7)
                    ldB(*nxt, &Vt[(size_t)(16 * (j + 1) + l15) * VSTR + half * 16]);
                o[j] = wmma_bf16(pa, *cur, o[j]);
            }
        }
    }

    // Normalize and write Y (bf16)
    unsigned short* yb = ybuf + (size_t)b * NPIX * CI;
#pragma unroll
    for (int v = 0; v < 8; ++v) {
        float inv = 1.f / lrow[v];
        int row = q0 + 16 * wave + v + 8 * half;
#pragma unroll
        for (int j = 0; j < 8; ++j)
            yb[(size_t)row * CI + 16 * j + l15] = f2bfu(o[j][v] * inv);
    }
}

// ---------------------------------------------------------------------------
// Kernel 3: out = x + BN(Y @ w_out^T + b_out).  128 rows x 256 cols per block.
// Epilogue transposes through LDS (reusing the dead Y tile) so the C-major
// output plane is written with coalesced float4 stores along n.
// ---------------------------------------------------------------------------
#define TFS (BR + 4)    // f32 transpose-buffer stride (132)

__global__ __launch_bounds__(256) void outproj_kernel(
    const unsigned short* __restrict__ ybuf,
    const unsigned short* __restrict__ w_out, const float* __restrict__ b_out,
    const float* __restrict__ gamma, const float* __restrict__ beta,
    const float* __restrict__ mean,  const float* __restrict__ var,
    const float* __restrict__ x, float* __restrict__ out)
{
    __shared__ union {
        unsigned short ys[BR * (CI + 8)];     // 34816 B
        float          tf[64 * TFS];          // 33792 B
    } sm;

    const int b    = blockIdx.y;
    const int n0   = blockIdx.x * BR;
    const int tid  = threadIdx.x;
    const int wave = tid >> 5, lane = tid & 31;
    const int half = lane >> 4, l15 = lane & 15;

    const unsigned short* yg = ybuf + (size_t)b * NPIX * CI;
#if USE_TDM
    if (wave == 0) {
        tdm_load_rows(0u, yg + (size_t)n0 * CI, BR);
        __builtin_amdgcn_s_wait_tensorcnt(0);
    }
    __syncthreads();
#else
    for (int i = 0; i < 32; ++i) {
        int e = i * 256 + tid; int pc = e & 63, r = e >> 6;
        *(unsigned int*)&sm.ys[r * (CI + 8) + 2 * pc] =
            ((const unsigned int*)yg)[(size_t)(n0 + r) * 64 + pc];
    }
    __syncthreads();
#endif

    v8f acc[16];
    v8f zero8 = {0.f,0.f,0.f,0.f,0.f,0.f,0.f,0.f};
#pragma unroll
    for (int j = 0; j < 16; ++j) acc[j] = zero8;

    const unsigned short* wrow = w_out + (size_t)l15 * CI + half * 16;
    for (int kc = 0; kc < CI; kc += 32) {
        Frag16 a; ldA(a, &sm.ys[(16 * wave + l15) * (CI + 8)], kc, half);
        Frag16 b0, b1;
        ldB(b0, wrow + kc);
#pragma unroll
        for (int j = 0; j < 16; ++j) {
            Frag16* cur = (j & 1) ? &b1 : &b0;
            Frag16* nxt = (j & 1) ? &b0 : &b1;
            if (j < 15) ldB(*nxt, wrow + (size_t)(16 * (j + 1)) * CI + kc);
            acc[j] = wmma_bf16(a, *cur, acc[j]);
        }
    }

    // Epilogue: 4 groups of 64 output channels; transpose via LDS, then
    // coalesced float4 residual-add + BN + store along n.
    for (int g = 0; g < 4; ++g) {
        __syncthreads();                       // ys / previous tf readers done
#pragma unroll
        for (int jj = 0; jj < 4; ++jj) {
            int j = 4 * g + jj;
            int cl = 16 * jj + l15;            // 0..63 within group
#pragma unroll
            for (int v = 0; v < 8; ++v)
                sm.tf[cl * TFS + 16 * wave + v + 8 * half] = acc[j][v];
        }
        __syncthreads();
        for (int i = 0; i < 8; ++i) {          // 64 cols x 32 float4 rows
            int e = i * 256 + tid;
            int qq = e & 31, cl = e >> 5;
            int co = g * 64 + cl;
            float inv   = gamma[co] * rsqrtf(var[co] + 1e-5f);
            float shift = beta[co] - mean[co] * inv + b_out[co] * inv;
            const float4 tv = *(const float4*)&sm.tf[cl * TFS + 4 * qq];
            size_t idx = ((size_t)b * C_ + co) * (size_t)NPIX + n0 + 4 * qq;
            const float4 xv = *(const float4*)(x + idx);
            float4 r;
            r.x = xv.x + inv * tv.x + shift;
            r.y = xv.y + inv * tv.y + shift;
            r.z = xv.z + inv * tv.z + shift;
            r.w = xv.w + inv * tv.w + shift;
            *(float4*)(out + idx) = r;
        }
    }
}

// ---------------------------------------------------------------------------
extern "C" void kernel_launch(void* const* d_in, const int* in_sizes, int n_in,
                              void* d_out, int out_size, void* d_ws, size_t ws_size,
                              hipStream_t stream)
{
    const float* x     = (const float*)d_in[0];
    const float* w_g   = (const float*)d_in[1];
    const float* b_g   = (const float*)d_in[2];
    const float* w_th  = (const float*)d_in[3];
    const float* b_th  = (const float*)d_in[4];
    const float* w_ph  = (const float*)d_in[5];
    const float* b_ph  = (const float*)d_in[6];
    const float* w_out = (const float*)d_in[7];
    const float* b_out = (const float*)d_in[8];
    const float* gamma = (const float*)d_in[9];
    const float* beta  = (const float*)d_in[10];
    const float* mean  = (const float*)d_in[11];
    const float* var   = (const float*)d_in[12];
    float* out = (float*)d_out;

    const size_t bufElems = (size_t)BATCH * NPIX * CI;     // 4 Mi bf16 = 8 MiB
    unsigned short* qbuf = (unsigned short*)d_ws;
    unsigned short* kbuf = qbuf + bufElems;
    unsigned short* vbuf = kbuf + bufElems;
    unsigned short* ybuf = vbuf + bufElems;
    unsigned short* wbuf = ybuf + bufElems;                // 131072 bf16 weights
    unsigned short* wthb = wbuf;                           // theta
    unsigned short* wphb = wbuf + 32768;                   // phi
    unsigned short* wgb  = wbuf + 65536;                   // g
    unsigned short* wob  = wbuf + 98304;                   // out

    cvtw_kernel<<<dim3(256), 256, 0, stream>>>(w_th, w_ph, w_g, w_out,
                                               (unsigned int*)wbuf);
    proj_kernel<<<dim3(NPIX / 64, BATCH), 128, 0, stream>>>(
        x, wthb, b_th, wphb, b_ph, wgb, b_g, qbuf, kbuf, vbuf);
    attn_kernel<<<dim3(NPIX / BR, BATCH), 256, 0, stream>>>(
        qbuf, kbuf, vbuf, ybuf);
    outproj_kernel<<<dim3(NPIX / BR, BATCH), 256, 0, stream>>>(
        ybuf, wob, b_out, gamma, beta, mean, var, x, out);
}
